// PointNet2LoRa_89258010346077
// MI455X (gfx1250) — compile-verified, matrix operand order checked by reading, not run on hardware
//
#include <hip/hip_runtime.h>
#include <math.h>

// ---------------------------------------------------------------------------
// PointNet++ (LoRA) inference for MI455X / gfx1250, wave32 + WMMA f16->f32.
// LoRA is folded into weights on-device; every linear layer is a WMMA GEMM.
// SA modules are fused per destination point (feature build + 3-layer MLP +
// masked max all in LDS) so the [m,64,C] message tensors never hit HBM.
// Row-wise MLPs stage activation tiles in LDS (f16) and keep WMMA accumulator
// fragments in VGPRs across K chunks.
// ---------------------------------------------------------------------------

typedef __attribute__((ext_vector_type(16))) _Float16 v16h;
typedef __attribute__((ext_vector_type(8)))  float    v8f;

#define WAVES 8          // 256 threads per block
#define KNEIGH 64
#define KC 128           // K-chunk (halves) staged in LDS by mlp_rows

// K index mapping for the 16-bit 16x32 A/B fragment layout (ISA 7.12.2):
// VGPR0..3 hold K=0..7 (lanes 0-15) / K=8..15 (lanes 16-31),
// VGPR4..7 hold K=16..23 / K=24..31.  h = half index 0..15, hi = lane>>4.
// Consecutive h within each group of 8 map to consecutive K, so the per-lane
// fragment loads coalesce into b128 accesses.
__device__ __forceinline__ int kmap16(int h, int hi) {
  int vg = h >> 1, p = h & 1;
  int base = (vg < 4) ? (vg * 2 + p) : (16 + (vg - 4) * 2 + p);
  return base + hi * 8;
}

// ---------------------------------------------------------------------------
// Fold LoRA: Weff[o][k] = W[o][k] + 2 * sum_r B[o][r]*A[r][k], stored f16,
// zero-padded to (coutPad x cinPad).  Bias padded to coutPad (f32).
// ---------------------------------------------------------------------------
__global__ void fold_lora_kernel(const float* __restrict__ W,
                                 const float* __restrict__ Amat,
                                 const float* __restrict__ Bmat,
                                 const float* __restrict__ bias,
                                 _Float16* __restrict__ Wout,
                                 float* __restrict__ bout,
                                 int cout, int cin, int cinPad) {
  int o = blockIdx.x;  // 0..coutPad-1
  for (int k = threadIdx.x; k < cinPad; k += blockDim.x) {
    float v = 0.f;
    if (o < cout && k < cin) {
      v = W[(size_t)o * cin + k];
#pragma unroll
      for (int r = 0; r < 8; ++r)
        v += 2.0f * Bmat[o * 8 + r] * Amat[(size_t)r * cin + k];
    }
    Wout[(size_t)o * cinPad + k] = (_Float16)v;
  }
  if (threadIdx.x == 0) bout[o] = (o < cout) ? bias[o] : 0.f;
}

// ---------------------------------------------------------------------------
// Farthest point sampling, one block per batch (matches jax scan semantics:
// emit last, min-update distances, argmax -> next).  Also gathers dst pos.
// ---------------------------------------------------------------------------
__global__ void fps_kernel(const float* __restrict__ pos, int N, int m,
                           int* __restrict__ idx_out, float* __restrict__ posd) {
  int b = blockIdx.x;
  pos += (size_t)b * N * 3; idx_out += (size_t)b * m; posd += (size_t)b * m * 3;
  __shared__ float sdist[4096];
  __shared__ float rv[256];
  __shared__ int   ri[256];
  __shared__ int   s_last;
  int tid = threadIdx.x;
  for (int i = tid; i < N; i += 256) sdist[i] = 3.0e38f;
  if (tid == 0) s_last = 0;
  __syncthreads();
  for (int it = 0; it < m; ++it) {
    int last = s_last;
    float lx = pos[last * 3 + 0], ly = pos[last * 3 + 1], lz = pos[last * 3 + 2];
    if (tid == 0) {
      idx_out[it] = last;
      posd[it * 3 + 0] = lx; posd[it * 3 + 1] = ly; posd[it * 3 + 2] = lz;
    }
    float bv = -1.f; int bi = 0;
    for (int i = tid; i < N; i += 256) {
      float dx = pos[i * 3 + 0] - lx, dy = pos[i * 3 + 1] - ly, dz = pos[i * 3 + 2] - lz;
      float nd = fminf(sdist[i], dx * dx + dy * dy + dz * dz);
      sdist[i] = nd;
      if (nd > bv) { bv = nd; bi = i; }
    }
    rv[tid] = bv; ri[tid] = bi;
    __syncthreads();
    for (int s = 128; s > 0; s >>= 1) {
      if (tid < s) {
        bool take = (rv[tid + s] > rv[tid]) ||
                    (rv[tid + s] == rv[tid] && ri[tid + s] < ri[tid]);
        if (take) { rv[tid] = rv[tid + s]; ri[tid] = ri[tid + s]; }
      }
      __syncthreads();
    }
    if (tid == 0) s_last = ri[0];
    __syncthreads();
  }
}

// ---------------------------------------------------------------------------
// Radius-ball top-64 (smallest d2 within r2; -1 marks invalid slots).
// One block per (dst point, batch).  Iterative argmin over LDS distances.
// ---------------------------------------------------------------------------
__global__ void radius_topk_kernel(const float* __restrict__ pos_s, int N,
                                   const float* __restrict__ pos_d, int m,
                                   float r2, int* __restrict__ nb) {
  int p = blockIdx.x, b = blockIdx.y;
  pos_s += (size_t)b * N * 3;
  pos_d += ((size_t)b * m + p) * 3;
  nb    += ((size_t)b * m + p) * KNEIGH;
  __shared__ float sd[4096];
  __shared__ float rv[256];
  __shared__ int   ri[256];
  int tid = threadIdx.x;
  float qx = pos_d[0], qy = pos_d[1], qz = pos_d[2];
  for (int i = tid; i < N; i += 256) {
    float dx = qx - pos_s[i * 3 + 0], dy = qy - pos_s[i * 3 + 1], dz = qz - pos_s[i * 3 + 2];
    sd[i] = dx * dx + dy * dy + dz * dz;
  }
  __syncthreads();
  for (int kth = 0; kth < KNEIGH; ++kth) {
    float bv = 3.0e38f; int bi = -1;
    for (int i = tid; i < N; i += 256)
      if (sd[i] < bv) { bv = sd[i]; bi = i; }
    rv[tid] = bv; ri[tid] = bi;
    __syncthreads();
    for (int s = 128; s > 0; s >>= 1) {
      if (tid < s) {
        bool take = (rv[tid + s] < rv[tid]) ||
                    (rv[tid + s] == rv[tid] && ri[tid + s] >= 0 && ri[tid + s] < ri[tid]);
        if (take) { rv[tid] = rv[tid + s]; ri[tid] = ri[tid + s]; }
      }
      __syncthreads();
    }
    if (tid == 0) {
      if (rv[0] <= r2 && ri[0] >= 0) { nb[kth] = ri[0]; sd[ri[0]] = 3.0e38f; }
      else nb[kth] = -1;
    }
    __syncthreads();
  }
}

// ---------------------------------------------------------------------------
// One WMMA GEMM layer over a fixed 64-row tile living in LDS (f16 in/out).
// 8 waves round-robin the 4 x (coutPad/16) output tiles.
// ---------------------------------------------------------------------------
__device__ __forceinline__ void wave_gemm_lds(const _Float16* __restrict__ Wg,
                                              const float* __restrict__ bg,
                                              const _Float16* in, int cinPad,
                                              _Float16* out, int coutPad,
                                              bool relu) {
  int wave = threadIdx.x >> 5, lane = threadIdx.x & 31;
  int lo = lane & 15, hi = lane >> 4;
  int nt = coutPad >> 4;
  for (int t = wave; t < 4 * nt; t += WAVES) {
    int mi = (t & 3) * 16;
    int ni = (t >> 2) * 16;
    v8f acc = {};
    for (int k0 = 0; k0 < cinPad; k0 += 32) {
      v16h a, bm;
#pragma unroll
      for (int h = 0; h < 16; ++h) {
        int k = k0 + kmap16(h, hi);
        a[h]  = in[(mi + lo) * cinPad + k];
        bm[h] = Wg[(size_t)(ni + lo) * cinPad + k];
      }
      acc = __builtin_amdgcn_wmma_f32_16x16x32_f16(false, a, false, bm,
                                                   (short)0, acc, false, false);
    }
    float bias = bg[ni + lo];
#pragma unroll
    for (int v = 0; v < 8; ++v) {
      float y = acc[v] + bias;
      if (relu) y = fmaxf(y, 0.f);
      out[(mi + v + hi * 8) * coutPad + (ni + lo)] = (_Float16)y;
    }
  }
}

// ---------------------------------------------------------------------------
// Fused SA module conv: one block per (dst point, batch).
// Build [64 x cin] neighbor features in LDS, run 3 WMMA layers, masked max.
// ---------------------------------------------------------------------------
__global__ __launch_bounds__(256)
void sa_conv_kernel(const float* __restrict__ xs, int cx,
                    const float* __restrict__ pos_s, int Ns,
                    const int* __restrict__ nb,
                    const float* __restrict__ pos_d, int m,
                    const _Float16* W0, const float* b0, int ci0, int co0,
                    const _Float16* W1, const float* b1, int ci1, int co1,
                    const _Float16* W2, const float* b2, int ci2, int co2,
                    float* __restrict__ out) {
  int p = blockIdx.x, b = blockIdx.y;
  xs    += (size_t)b * Ns * cx;
  pos_s += (size_t)b * Ns * 3;
  nb    += ((size_t)b * m + p) * KNEIGH;
  pos_d += ((size_t)b * m + p) * 3;
  out   += ((size_t)b * m + p) * co2;

  __shared__ _Float16 featS[64 * 160];   // max cinPad = 160 (sa2)
  __shared__ _Float16 bufA[64 * 256];    // max coutPad = 256
  __shared__ _Float16 bufB[64 * 256];
  __shared__ int nbS[KNEIGH];

  int tid = threadIdx.x;
  if (tid < KNEIGH) nbS[tid] = nb[tid];
  __syncthreads();
  float qx = pos_d[0], qy = pos_d[1], qz = pos_d[2];

  for (int e = tid; e < 64 * ci0; e += 256) {
    int row = e / ci0, c = e - row * ci0;
    int j = nbS[row];
    float v = 0.f;
    if (j >= 0) {
      if (c < cx)            v = xs[(size_t)j * cx + c];
      else if (c == cx)      v = pos_s[j * 3 + 0] - qx;
      else if (c == cx + 1)  v = pos_s[j * 3 + 1] - qy;
      else if (c == cx + 2)  v = pos_s[j * 3 + 2] - qz;
    }
    featS[e] = (_Float16)v;
  }
  __syncthreads();

  wave_gemm_lds(W0, b0, featS, ci0, bufA, co0, true);
  __syncthreads();
  wave_gemm_lds(W1, b1, bufA, ci1, bufB, co1, true);
  __syncthreads();
  wave_gemm_lds(W2, b2, bufB, ci2, bufA, co2, false);
  __syncthreads();

  for (int c = tid; c < co2; c += 256) {
    float mx = -3.0e38f;
#pragma unroll 4
    for (int r = 0; r < KNEIGH; ++r)
      if (nbS[r] >= 0) mx = fmaxf(mx, (float)bufA[r * co2 + c]);
    out[c] = mx;
  }
}

// ---------------------------------------------------------------------------
// Generic row-wise WMMA MLP layer: Y[M,coutPad] = act(concat(A,B) @ W^T + b).
// Block computes a 64-row x 128-col output tile.  The 64 x KC activation
// chunk is cooperatively staged into LDS as f16 once per K-chunk, so global
// activation traffic is exactly one pass per (M-block, N-block); accumulator
// fragments stay in VGPRs across chunks.  srcA may be broadcast (x3).
// ---------------------------------------------------------------------------
__global__ __launch_bounds__(256)
void mlp_rows_kernel(const float* __restrict__ srcA, int ca, int aBcast,
                     const float* __restrict__ srcB, int cb,
                     const _Float16* __restrict__ Wg, const float* __restrict__ bg,
                     int cinPad, int coutPad,
                     float* __restrict__ dst, int M, int relu) {
  int b  = blockIdx.z;
  int m0 = blockIdx.x * 64;
  int n0 = blockIdx.y * 128;
  srcA += (size_t)b * (aBcast ? (size_t)ca : (size_t)M * ca);
  if (srcB) srcB += (size_t)b * M * cb;
  dst += (size_t)b * M * coutPad;

  __shared__ _Float16 aS[64 * KC];   // 16 KB

  int tid  = threadIdx.x;
  int wave = tid >> 5, lane = tid & 31;
  int lo = lane & 15, hi = lane >> 4;

  int nLocal = coutPad - n0; if (nLocal > 128) nLocal = 128;
  int ntiles = (nLocal >> 4) * 4;          // (m,n) tiles in this block, <=32

  v8f acc0 = {}, acc1 = {}, acc2 = {}, acc3 = {};

  for (int k0 = 0; k0 < cinPad; k0 += KC) {
    int kc = cinPad - k0; if (kc > KC) kc = KC;
    // cooperative stage of the A chunk (f32 -> f16), zero padded
    for (int e = tid; e < 64 * KC; e += 256) {
      int r = e >> 7;          // row within tile (KC == 128)
      int c = e & (KC - 1);
      float v = 0.f;
      int row = m0 + r;
      int k = k0 + c;
      if (row < M && c < kc) {
        int ar = aBcast ? 0 : row;
        if (k < ca)                    v = srcA[(size_t)ar * ca + k];
        else if (srcB && k < ca + cb)  v = srcB[(size_t)row * cb + (k - ca)];
      }
      aS[e] = (_Float16)v;
    }
    __syncthreads();
#pragma unroll
    for (int ai = 0; ai < 4; ++ai) {
      int t = wave + ai * WAVES;
      if (t < ntiles) {
        int mi = (t & 3) * 16;
        int ni = n0 + (t >> 2) * 16;
        v8f a8 = (ai == 0) ? acc0 : (ai == 1) ? acc1 : (ai == 2) ? acc2 : acc3;
        for (int kk = 0; kk < kc; kk += 32) {
          v16h av, bv;
#pragma unroll
          for (int h = 0; h < 16; ++h) {
            int km = kmap16(h, hi);
            av[h] = aS[(mi + lo) * KC + kk + km];
            bv[h] = Wg[(size_t)(ni + lo) * cinPad + k0 + kk + km];
          }
          a8 = __builtin_amdgcn_wmma_f32_16x16x32_f16(false, av, false, bv,
                                                      (short)0, a8, false, false);
        }
        if (ai == 0) acc0 = a8; else if (ai == 1) acc1 = a8;
        else if (ai == 2) acc2 = a8; else acc3 = a8;
      }
    }
    __syncthreads();
  }

  // epilogue: bias + activation + store
#pragma unroll
  for (int ai = 0; ai < 4; ++ai) {
    int t = wave + ai * WAVES;
    if (t < ntiles) {
      int mi = m0 + (t & 3) * 16;
      int ni = n0 + (t >> 2) * 16;
      v8f a8 = (ai == 0) ? acc0 : (ai == 1) ? acc1 : (ai == 2) ? acc2 : acc3;
      float bias = bg[ni + lo];
#pragma unroll
      for (int v = 0; v < 8; ++v) {
        int row = mi + v + hi * 8;
        if (row < M) {
          float y = a8[v] + bias;
          if (relu) y = fmaxf(y, 0.f);
          dst[(size_t)row * coutPad + ni + lo] = y;
        }
      }
    }
  }
}

// ---------------------------------------------------------------------------
// Column-wise max pool over M rows (global SA pooling).
// ---------------------------------------------------------------------------
__global__ void maxpool_rows_kernel(const float* __restrict__ in, int M, int C,
                                    float* __restrict__ out) {
  int b = blockIdx.y;
  int c = blockIdx.x * blockDim.x + threadIdx.x;
  if (c >= C) return;
  in += (size_t)b * M * C;
  float mx = -3.0e38f;
  for (int r = 0; r < M; ++r) mx = fmaxf(mx, in[(size_t)r * C + c]);
  out[(size_t)b * C + c] = mx;
}

// ---------------------------------------------------------------------------
// knn(k=3) inverse-squared-distance interpolation, one block per dst point.
// ---------------------------------------------------------------------------
__global__ void knn3_interp_kernel(const float* __restrict__ xsrc, int C,
                                   const float* __restrict__ pos_s, int Ns,
                                   const float* __restrict__ pos_d, int Md,
                                   float* __restrict__ out) {
  int p = blockIdx.x, b = blockIdx.y;
  xsrc  += (size_t)b * Ns * C;
  pos_s += (size_t)b * Ns * 3;
  pos_d += ((size_t)b * Md + p) * 3;
  out   += ((size_t)b * Md + p) * C;
  __shared__ float sd[832];
  __shared__ float rv[128];
  __shared__ int   ri[128];
  __shared__ int   ci[3];
  __shared__ float cw[3];
  int tid = threadIdx.x;
  float qx = pos_d[0], qy = pos_d[1], qz = pos_d[2];
  for (int i = tid; i < Ns; i += 128) {
    float dx = qx - pos_s[i * 3 + 0], dy = qy - pos_s[i * 3 + 1], dz = qz - pos_s[i * 3 + 2];
    sd[i] = dx * dx + dy * dy + dz * dz;
  }
  __syncthreads();
  for (int kth = 0; kth < 3; ++kth) {
    float bv = 3.0e38f; int bi = -1;
    for (int i = tid; i < Ns; i += 128)
      if (sd[i] < bv) { bv = sd[i]; bi = i; }
    rv[tid] = bv; ri[tid] = bi;
    __syncthreads();
    for (int s = 64; s > 0; s >>= 1) {
      if (tid < s) {
        bool take = (rv[tid + s] < rv[tid]) ||
                    (rv[tid + s] == rv[tid] && ri[tid + s] >= 0 && ri[tid + s] < ri[tid]);
        if (take) { rv[tid] = rv[tid + s]; ri[tid] = ri[tid + s]; }
      }
      __syncthreads();
    }
    if (tid == 0) {
      int sel = ri[0] < 0 ? 0 : ri[0];
      ci[kth] = sel;
      cw[kth] = 1.0f / fmaxf(rv[0], 1e-16f);
      sd[sel] = 3.0e38f;
    }
    __syncthreads();
  }
  float wsum = cw[0] + cw[1] + cw[2];
  for (int c = tid; c < C; c += 128) {
    float acc = cw[0] * xsrc[(size_t)ci[0] * C + c] +
                cw[1] * xsrc[(size_t)ci[1] * C + c] +
                cw[2] * xsrc[(size_t)ci[2] * C + c];
    out[c] = acc / wsum;
  }
}

// ---------------------------------------------------------------------------
// log_softmax over 13 classes (input stride 16 due to cout padding).
// ---------------------------------------------------------------------------
__global__ void log_softmax13_kernel(const float* __restrict__ in,
                                     float* __restrict__ out, int Mtot) {
  int i = blockIdx.x * blockDim.x + threadIdx.x;
  if (i >= Mtot) return;
  const float* r = in + (size_t)i * 16;
  float mx = -3.0e38f;
#pragma unroll
  for (int c = 0; c < 13; ++c) mx = fmaxf(mx, r[c]);
  float s = 0.f;
#pragma unroll
  for (int c = 0; c < 13; ++c) s += expf(r[c] - mx);
  float l = logf(s);
#pragma unroll
  for (int c = 0; c < 13; ++c) out[(size_t)i * 13 + c] = r[c] - mx - l;
}

// ---------------------------------------------------------------------------
// Host orchestration
// ---------------------------------------------------------------------------
static inline int pad32(int v) { return (v + 31) & ~31; }
static inline int pad16(int v) { return (v + 15) & ~15; }

extern "C" void kernel_launch(void* const* d_in, const int* in_sizes, int n_in,
                              void* d_out, int out_size, void* d_ws, size_t ws_size,
                              hipStream_t stream) {
  (void)in_sizes; (void)n_in; (void)out_size; (void)ws_size;
  const int B = 8, N = 4096, M1 = 820, M2 = 205;

  const float* x   = (const float*)d_in[0];
  const float* pos = (const float*)d_in[1];

  // Param pytree order assumed: top dict {x,pos,params} in insertion order;
  // params flattened with sorted keys: fp1,fp2,fp3,head,sa1,sa2,sa3; each a
  // list of layers; each layer dict sorted: A,B,W,b.
  struct LC { int cin, cout; };
  static const int nl[7] = {3, 2, 2, 3, 3, 3, 3};
  static const LC cfg[7][3] = {
      {{134, 128}, {128, 128}, {128, 128}},    // fp1
      {{384, 256}, {256, 128}, {0, 0}},        // fp2
      {{1280, 256}, {256, 256}, {0, 0}},       // fp3
      {{128, 128}, {128, 128}, {128, 13}},     // head
      {{9, 64}, {64, 64}, {64, 128}},          // sa1
      {{131, 128}, {128, 128}, {128, 256}},    // sa2
      {{259, 256}, {256, 512}, {512, 1024}},   // sa3
  };
  const float *A_[19], *Bm_[19], *W_[19], *bb_[19];
  int cin_[19], cout_[19], cinP_[19], coutP_[19];
  {
    int li = 0, di = 2;
    for (int g = 0; g < 7; ++g)
      for (int l = 0; l < nl[g]; ++l) {
        A_[li]  = (const float*)d_in[di + 0];
        Bm_[li] = (const float*)d_in[di + 1];
        W_[li]  = (const float*)d_in[di + 2];
        bb_[li] = (const float*)d_in[di + 3];
        di += 4;
        cin_[li] = cfg[g][l].cin;   cout_[li] = cfg[g][l].cout;
        cinP_[li] = pad32(cin_[li]); coutP_[li] = pad16(cout_[li]);
        ++li;
      }
  }
  // layer index map: fp1=0..2 fp2=3..4 fp3=5..6 head=7..9 sa1=10..12 sa2=13..15 sa3=16..18

  // Workspace cursor allocation
  char* wsb = (char*)d_ws;
  size_t cur = 0;
  auto alloc = [&](size_t bytes) -> void* {
    void* p = wsb + cur;
    cur += (bytes + 255) & ~(size_t)255;
    return p;
  };

  _Float16* Wf[19];
  float*    bf[19];
  for (int l = 0; l < 19; ++l) {
    Wf[l] = (_Float16*)alloc((size_t)coutP_[l] * cinP_[l] * sizeof(_Float16));
    bf[l] = (float*)alloc((size_t)coutP_[l] * sizeof(float));
  }
  int*   idx1 = (int*)alloc((size_t)B * M1 * sizeof(int));
  float* pos1 = (float*)alloc((size_t)B * M1 * 3 * sizeof(float));
  int*   idx2 = (int*)alloc((size_t)B * M2 * sizeof(int));
  float* pos2 = (float*)alloc((size_t)B * M2 * 3 * sizeof(float));
  int*   nb1  = (int*)alloc((size_t)B * M1 * KNEIGH * sizeof(int));
  int*   nb2  = (int*)alloc((size_t)B * M2 * KNEIGH * sizeof(int));
  float* x1   = (float*)alloc((size_t)B * M1 * 128 * sizeof(float));
  float* x2   = (float*)alloc((size_t)B * M2 * 256 * sizeof(float));
  float* x3f  = (float*)alloc((size_t)B * M2 * 1024 * sizeof(float));
  float* x3   = (float*)alloc((size_t)B * 1024 * sizeof(float));
  float* f3   = (float*)alloc((size_t)B * M2 * 256 * sizeof(float));
  float* xi2  = (float*)alloc((size_t)B * M1 * 256 * sizeof(float));
  float* f2   = (float*)alloc((size_t)B * M1 * 128 * sizeof(float));
  float* xi1  = (float*)alloc((size_t)B * N * 128 * sizeof(float));
  float* f1   = (float*)alloc((size_t)B * N * 128 * sizeof(float));
  float* t0   = (float*)alloc((size_t)B * N * 128 * sizeof(float));
  float* t1   = (float*)alloc((size_t)B * N * 128 * sizeof(float));
  float* hout = (float*)alloc((size_t)B * N * 16 * sizeof(float));

  // 1) Fold LoRA into f16 weights (weights then live in the 192MB L2)
  for (int l = 0; l < 19; ++l)
    fold_lora_kernel<<<dim3(coutP_[l]), 128, 0, stream>>>(
        W_[l], A_[l], Bm_[l], bb_[l], Wf[l], bf[l], cout_[l], cin_[l], cinP_[l]);

  // helper for mlp_rows grids
  auto mgrid = [](int M, int coutPad, int Bb) {
    return dim3((unsigned)((M + 63) / 64), (unsigned)((coutPad + 127) / 128), (unsigned)Bb);
  };

  // 2) SA1: FPS(0.2) -> radius(0.2) -> fused WMMA conv, out x1 [B,820,128]
  fps_kernel<<<dim3(B), 256, 0, stream>>>(pos, N, M1, idx1, pos1);
  radius_topk_kernel<<<dim3(M1, B), 256, 0, stream>>>(pos, N, pos1, M1, 0.04f, nb1);
  sa_conv_kernel<<<dim3(M1, B), 256, 0, stream>>>(
      x, 6, pos, N, nb1, pos1, M1,
      Wf[10], bf[10], cinP_[10], coutP_[10],
      Wf[11], bf[11], cinP_[11], coutP_[11],
      Wf[12], bf[12], cinP_[12], coutP_[12], x1);

  // 3) SA2: FPS(0.25) -> radius(0.4) -> fused conv, out x2 [B,205,256]
  fps_kernel<<<dim3(B), 256, 0, stream>>>(pos1, M1, M2, idx2, pos2);
  radius_topk_kernel<<<dim3(M2, B), 256, 0, stream>>>(pos1, M1, pos2, M2, 0.16f, nb2);
  sa_conv_kernel<<<dim3(M2, B), 256, 0, stream>>>(
      x1, 128, pos1, M1, nb2, pos2, M2,
      Wf[13], bf[13], cinP_[13], coutP_[13],
      Wf[14], bf[14], cinP_[14], coutP_[14],
      Wf[15], bf[15], cinP_[15], coutP_[15], x2);

  // 4) Global SA: MLP(concat(x2,pos2)) -> max pool -> x3 [B,1024]
  mlp_rows_kernel<<<mgrid(M2, coutP_[16], B), 256, 0, stream>>>(x2, 256, 0, pos2, 3,
      Wf[16], bf[16], cinP_[16], coutP_[16], t0, M2, 1);
  mlp_rows_kernel<<<mgrid(M2, coutP_[17], B), 256, 0, stream>>>(t0, 256, 0, nullptr, 0,
      Wf[17], bf[17], cinP_[17], coutP_[17], t1, M2, 1);
  mlp_rows_kernel<<<mgrid(M2, coutP_[18], B), 256, 0, stream>>>(t1, 512, 0, nullptr, 0,
      Wf[18], bf[18], cinP_[18], coutP_[18], x3f, M2, 0);
  maxpool_rows_kernel<<<dim3(4, B), 256, 0, stream>>>(x3f, M2, 1024, x3);

  // 5) FP3: concat(broadcast x3, x2) -> 256 relu -> 256
  mlp_rows_kernel<<<mgrid(M2, coutP_[5], B), 256, 0, stream>>>(x3, 1024, 1, x2, 256,
      Wf[5], bf[5], cinP_[5], coutP_[5], t0, M2, 1);
  mlp_rows_kernel<<<mgrid(M2, coutP_[6], B), 256, 0, stream>>>(t0, 256, 0, nullptr, 0,
      Wf[6], bf[6], cinP_[6], coutP_[6], f3, M2, 0);

  // 6) FP2: knn3(f3 @ pos2 -> pos1), concat(xi2, x1) -> 256 relu -> 128
  knn3_interp_kernel<<<dim3(M1, B), 128, 0, stream>>>(f3, 256, pos2, M2, pos1, M1, xi2);
  mlp_rows_kernel<<<mgrid(M1, coutP_[3], B), 256, 0, stream>>>(xi2, 256, 0, x1, 128,
      Wf[3], bf[3], cinP_[3], coutP_[3], t0, M1, 1);
  mlp_rows_kernel<<<mgrid(M1, coutP_[4], B), 256, 0, stream>>>(t0, 256, 0, nullptr, 0,
      Wf[4], bf[4], cinP_[4], coutP_[4], f2, M1, 0);

  // 7) FP1: knn3(f2 @ pos1 -> pos), concat(xi1, x) -> 128 relu x2 -> 128
  knn3_interp_kernel<<<dim3(N, B), 128, 0, stream>>>(f2, 128, pos1, M1, pos, N, xi1);
  mlp_rows_kernel<<<mgrid(N, coutP_[0], B), 256, 0, stream>>>(xi1, 128, 0, x, 6,
      Wf[0], bf[0], cinP_[0], coutP_[0], t0, N, 1);
  mlp_rows_kernel<<<mgrid(N, coutP_[1], B), 256, 0, stream>>>(t0, 128, 0, nullptr, 0,
      Wf[1], bf[1], cinP_[1], coutP_[1], t1, N, 1);
  mlp_rows_kernel<<<mgrid(N, coutP_[2], B), 256, 0, stream>>>(t1, 128, 0, nullptr, 0,
      Wf[2], bf[2], cinP_[2], coutP_[2], f1, N, 0);

  // 8) Head: 128 relu -> 128 relu -> 13 (padded 16) -> log_softmax
  mlp_rows_kernel<<<mgrid(N, coutP_[7], B), 256, 0, stream>>>(f1, 128, 0, nullptr, 0,
      Wf[7], bf[7], cinP_[7], coutP_[7], t0, N, 1);
  mlp_rows_kernel<<<mgrid(N, coutP_[8], B), 256, 0, stream>>>(t0, 128, 0, nullptr, 0,
      Wf[8], bf[8], cinP_[8], coutP_[8], t1, N, 1);
  mlp_rows_kernel<<<mgrid(N, coutP_[9], B), 256, 0, stream>>>(t1, 128, 0, nullptr, 0,
      Wf[9], bf[9], cinP_[9], coutP_[9], hout, N, 0);
  log_softmax13_kernel<<<dim3((B * N + 255) / 256), 256, 0, stream>>>(
      hout, (float*)d_out, B * N);
}
